// MHA_87729001988494
// MI455X (gfx1250) — matrix operand order repaired and checked by reading, not compiled, not run on hardware
//
#include <hip/hip_runtime.h>
#include <hip/hip_bf16.h>
#include <math.h>

// ---- problem constants (match reference) ----
#define B_  4
#define D_  1024
#define S_  2048
#define H_  16
#define HD_ 64
#define M_  (B_ * S_)   // 8192 rows for the projection GEMMs

typedef _Float16 v16h __attribute__((ext_vector_type(16)));
typedef _Float16 v8h  __attribute__((ext_vector_type(8)));
typedef _Float16 v4h  __attribute__((ext_vector_type(4)));
typedef float    v8f  __attribute__((ext_vector_type(8)));

// TDM path: probe-confirmed builtins (see CDNA5_HIP.md). Guarded so the host
// pass and TDM-less toolchains fall back to synchronous LDS staging.
#if defined(__gfx1250__) && __has_builtin(__builtin_amdgcn_tensor_load_to_lds) && \
    __has_builtin(__builtin_amdgcn_s_wait_tensorcnt)
#define USE_TDM 1
#else
#define USE_TDM 0
#endif

// Build a 16-half A/B fragment from two 16-byte segments (ds/global b128 loads).
__device__ __forceinline__ v16h ldfrag(const _Float16* p0, const _Float16* p1) {
    v8h lo = *(const v8h*)p0;
    v8h hi = *(const v8h*)p1;
    return __builtin_shufflevector(lo, hi, 0,1,2,3,4,5,6,7,8,9,10,11,12,13,14,15);
}

#if USE_TDM
typedef unsigned int u32x4 __attribute__((ext_vector_type(4)));
typedef int          i32x4 __attribute__((ext_vector_type(4)));
typedef int          i32x8 __attribute__((ext_vector_type(8)));

// Issue one TDM 2D tile copy: 128 rows x 32 f16 from a row-major [rows,1024]
// f16 tensor into LDS (rows contiguous, 64B per row). ISA 08_async_tensor §8.
__device__ __forceinline__ void tdm_load_tile(unsigned lds_off,
                                              const _Float16* gptr,
                                              unsigned tensor_rows) {
    unsigned long long ga = (unsigned long long)(size_t)gptr;  // byte address
    u32x4 g0;
    g0.x = 1u;                                   // count=1 (valid), user mode
    g0.y = lds_off;                              // lds_addr (bytes)
    g0.z = (unsigned)ga;                         // global_addr[31:0]
    g0.w = (unsigned)((ga >> 32) & 0x01FFFFFFu)  // global_addr[56:32]
         | 0x80000000u;                          // type=2 ("image")
    i32x8 g1;
    g1[0] = 0x00010000;                          // wg_mask=0, data_size=1 (2B)
    g1[1] = (int)(1024u << 16);                  // tensor_dim0 = 1024 (K)
    g1[2] = (int)(tensor_rows << 16);            // tensor_dim1 (rows)
    g1[3] = (int)(32u << 16);                    // tile_dim0 = 32
    g1[4] = 128;                                 // tile_dim1 = 128, tile_dim2=0
    g1[5] = 1024;                                // tensor_dim0_stride = 1024
    g1[6] = 0;
    g1[7] = 0;                                   // dim1_stride unused (2D tile)
    i32x4 z4 = {0, 0, 0, 0};
#if __clang_major__ >= 23
    i32x8 z8 = {0, 0, 0, 0, 0, 0, 0, 0};
    __builtin_amdgcn_tensor_load_to_lds(g0, g1, z4, z4, z8, 0);
#else
    __builtin_amdgcn_tensor_load_to_lds(g0, g1, z4, z4, 0);
#endif
}
#endif

// ---------------------------------------------------------------------------
// Kernel 0: weights fp32 -> f16 (done once; tiles then DMA directly to LDS)
// ---------------------------------------------------------------------------
__global__ __launch_bounds__(256)
void wcast_kernel(const float* __restrict__ w, _Float16* __restrict__ wh) {
    int i4 = (blockIdx.x * 256 + threadIdx.x) * 4;
    float4 f = *(const float4*)&w[i4];
    v4h h;
    h[0] = (_Float16)f.x; h[1] = (_Float16)f.y;
    h[2] = (_Float16)f.z; h[3] = (_Float16)f.w;
    *(v4h*)&wh[i4] = h;
}

// ---------------------------------------------------------------------------
// Kernel 1: x[B,D,S] + positional encoding -> Xh[B,S,D] (f16)
// ---------------------------------------------------------------------------
__global__ __launch_bounds__(256)
void pe_cast_kernel(const float* __restrict__ x, _Float16* __restrict__ xh) {
    int idx = blockIdx.x * 256 + threadIdx.x;     // linear over [b][s][d]
    int d = idx % D_;
    int s = (idx / D_) % S_;
    int b = idx / (D_ * S_);
    float v = x[((size_t)b * D_ + d) * S_ + s];
    int   dh   = d & ~1;
    // exp(-dh * ln(10000)/D); ln(1e4)/1024 = 0.0089944735f
    float freq = __expf(-(float)dh * 0.0089944735f);
    float ang  = (float)s * freq;
    float pe   = (d & 1) ? __cosf(ang) : __sinf(ang);
    xh[idx] = (_Float16)(v + pe);
}

// ---------------------------------------------------------------------------
// Kernel 2: WMMA GEMM  Out[m,n] = sum_k A[m,k] * Wh[n,k] + bias[n]
//   A, Wh: f16 row-major, K = 1024. Block tile 128x128, 8 waves (2x4),
//   wave tile 64x32, K-step 32. Tiles staged to LDS by the Tensor Data
//   Mover (wave 0 issues descriptors), double-buffered: DMA of step kb+1
//   overlaps wmma of step kb; one barrier + s_wait_tensorcnt per step.
// mode: 0=Q (scale 1/8, [B,H,S,64])  1=K ([B,H,S,64])
//       2=V transposed ([B,H,64,S])  3=out-proj (fp32 -> d_out [B,D,S])
// ---------------------------------------------------------------------------
__global__ __launch_bounds__(256)
void gemm_wmma_kernel(const _Float16* __restrict__ A, const _Float16* __restrict__ Wh,
                      const float* __restrict__ bias, _Float16* __restrict__ outH,
                      float* __restrict__ outF, int mode)
{
    __shared__ _Float16 lA[2][128 * 32];   // 2 x 8 KB
    __shared__ _Float16 lB[2][128 * 32];   // 2 x 8 KB

    const int t    = threadIdx.x;
    const int lane = t & 31;
    const int wid  = t >> 5;
    const int wm   = wid & 1;          // wave row   (2 x 64 rows)
    const int wn   = wid >> 1;         // wave col   (4 x 32 cols)
    const int lh   = lane >> 4;        // lane half
    const int ll   = lane & 15;
    const int mBase = blockIdx.y * 128;
    const int nBase = blockIdx.x * 128;

    v8f acc[4][2];
    #pragma unroll
    for (int i = 0; i < 4; ++i)
        #pragma unroll
        for (int j = 0; j < 2; ++j) acc[i][j] = {};

#if USE_TDM
    // Prologue: DMA the first K-step tiles, wait, publish to all waves.
    if (t < 32) {
        tdm_load_tile((unsigned)(size_t)&lA[0][0], A  + (size_t)mBase * D_, M_);
        tdm_load_tile((unsigned)(size_t)&lB[0][0], Wh + (size_t)nBase * D_, D_);
        __builtin_amdgcn_s_wait_tensorcnt(0);
    }
    __syncthreads();
#endif

    for (int kb = 0; kb < D_ / 32; ++kb) {
        const int cur = kb & 1;
#if USE_TDM
        // Prefetch next K-step via TDM while this step computes.
        if (kb + 1 < D_ / 32 && t < 32) {
            const int k1 = (kb + 1) * 32;
            tdm_load_tile((unsigned)(size_t)&lA[cur ^ 1][0],
                          A  + (size_t)mBase * D_ + k1, M_);
            tdm_load_tile((unsigned)(size_t)&lB[cur ^ 1][0],
                          Wh + (size_t)nBase * D_ + k1, D_);
        }
#else
        // Fallback: synchronous staging into buffer 0.
        const int k0 = kb * 32;
        #pragma unroll
        for (int p = 0; p < 2; ++p) {
            int seg = t + p * 256;             // 512 segments of 8 halves
            int row = seg >> 2;
            int c8  = (seg & 3) << 3;
            *(v8h*)&lA[0][row * 32 + c8] =
                *(const v8h*)&A[((size_t)(mBase + row)) * D_ + k0 + c8];
            *(v8h*)&lB[0][row * 32 + c8] =
                *(const v8h*)&Wh[((size_t)(nBase + row)) * D_ + k0 + c8];
        }
        __syncthreads();
#endif
        // ---- compute: 8 wmma per wave per K-step ----
        const _Float16* __restrict__ bufA = &lA[USE_TDM ? cur : 0][0];
        const _Float16* __restrict__ bufB = &lB[USE_TDM ? cur : 0][0];
        v16h bf[2];
        #pragma unroll
        for (int j = 0; j < 2; ++j) {
            // B frag: lane column n = ll, K segs at lh*8 and 16+lh*8
            const _Float16* bp = &bufB[(wn * 32 + j * 16 + ll) * 32 + lh * 8];
            bf[j] = ldfrag(bp, bp + 16);
        }
        #pragma unroll
        for (int i = 0; i < 4; ++i) {
            // A frag: lane row = ll, K segs at lh*8 and 16+lh*8
            const _Float16* ap = &bufA[(wm * 64 + i * 16 + ll) * 32 + lh * 8];
            v16h af = ldfrag(ap, ap + 16);
            #pragma unroll
            for (int j = 0; j < 2; ++j)
                acc[i][j] = __builtin_amdgcn_wmma_f32_16x16x32_f16(
                    false, af, false, bf[j], (short)0, acc[i][j], false, false);
        }
#if USE_TDM
        if (t < 32) __builtin_amdgcn_s_wait_tensorcnt(0);
        __syncthreads();
#else
        __syncthreads();
#endif
    }

    // ---- epilogue (C layout: VGPR r -> m = r + 8*laneHalf, n = ll) ----
    #pragma unroll
    for (int i = 0; i < 4; ++i) {
        #pragma unroll
        for (int j = 0; j < 2; ++j) {
            #pragma unroll
            for (int r = 0; r < 8; ++r) {
                int m = mBase + wm * 64 + i * 16 + r + lh * 8;
                int n = nBase + wn * 32 + j * 16 + ll;
                float v = acc[i][j][r] + bias[n];
                int b = m / S_, s = m % S_;
                if (mode == 0) {          // Q, softmax scale folded in
                    outH[(((size_t)b * H_ + (n >> 6)) * S_ + s) * HD_ + (n & 63)] =
                        (_Float16)(v * 0.125f);
                } else if (mode == 1) {   // K
                    outH[(((size_t)b * H_ + (n >> 6)) * S_ + s) * HD_ + (n & 63)] =
                        (_Float16)v;
                } else if (mode == 2) {   // V transposed for PV B-fragments
                    outH[(((size_t)b * H_ + (n >> 6)) * HD_ + (n & 63)) * S_ + s] =
                        (_Float16)v;
                } else {                  // out-proj: fp32, [B,D,S]
                    outF[((size_t)b * D_ + n) * S_ + s] = v;
                }
            }
        }
    }
}

// ---------------------------------------------------------------------------
// Kernel 3: flash attention per (b, h, 64-query tile). 4 waves; each wave
// owns 16 query rows. Online softmax; P converted C-layout -> A-layout via
// a per-wave LDS tile; V consumed from the [B,H,64,S] transposed buffer.
// ---------------------------------------------------------------------------
__global__ __launch_bounds__(128)
void attn_kernel(const _Float16* __restrict__ Q, const _Float16* __restrict__ K,
                 const _Float16* __restrict__ Vt, _Float16* __restrict__ O)
{
    __shared__ _Float16 lP[4][16 * 64];   // 2 KB per wave

    const int t    = threadIdx.x;
    const int lane = t & 31;
    const int w    = t >> 5;
    const int lh   = lane >> 4;
    const int ll   = lane & 15;
    const int h    = blockIdx.y, b = blockIdx.z;
    const int bh   = b * H_ + h;
    const _Float16* qptr = Q  + (size_t)bh * S_ * HD_;
    const _Float16* kptr = K  + (size_t)bh * S_ * HD_;
    const _Float16* vptr = Vt + (size_t)bh * HD_ * S_;
    const int q0 = blockIdx.x * 64 + w * 16;

    // Q fragments (held in registers for the whole loop)
    v16h qf[2];
    {
        const _Float16* qr = qptr + (size_t)(q0 + ll) * HD_;
        #pragma unroll
        for (int kk = 0; kk < 2; ++kk) {
            const _Float16* p = qr + kk * 32 + lh * 8;
            qf[kk] = ldfrag(p, p + 16);
        }
    }

    v8f oacc[4];
    #pragma unroll
    for (int j = 0; j < 4; ++j) oacc[j] = {};
    float mrun[8], lrun[8];
    #pragma unroll
    for (int r = 0; r < 8; ++r) { mrun[r] = -1e30f; lrun[r] = 0.f; }

    for (int kt = 0; kt < S_ / 64; ++kt) {
        const int key0 = kt * 64;

        // ---- scores: S = Q Kt (Q pre-scaled), 8 wmma ----
        v8f sacc[4];
        #pragma unroll
        for (int j = 0; j < 4; ++j) sacc[j] = {};
        #pragma unroll
        for (int j = 0; j < 4; ++j) {
            const _Float16* kr = kptr + (size_t)(key0 + j * 16 + ll) * HD_;
            #pragma unroll
            for (int kk = 0; kk < 2; ++kk) {
                const _Float16* p = kr + kk * 32 + lh * 8;
                v16h kf = ldfrag(p, p + 16);
                sacc[j] = __builtin_amdgcn_wmma_f32_16x16x32_f16(
                    false, qf[kk], false, kf, (short)0, sacc[j], false, false);
            }
        }

        // ---- online softmax (row reductions within 16-lane halves) ----
        #pragma unroll
        for (int r = 0; r < 8; ++r) {
            float tmax = fmaxf(fmaxf(sacc[0][r], sacc[1][r]),
                               fmaxf(sacc[2][r], sacc[3][r]));
            #pragma unroll
            for (int off = 1; off <= 8; off <<= 1)
                tmax = fmaxf(tmax, __shfl_xor(tmax, off, 32));
            float mnew  = fmaxf(mrun[r], tmax);
            float alpha = __expf(mrun[r] - mnew);
            float tsum  = 0.f;
            #pragma unroll
            for (int j = 0; j < 4; ++j) {
                float p = __expf(sacc[j][r] - mnew);
                sacc[j][r] = p;
                tsum += p;
            }
            #pragma unroll
            for (int off = 1; off <= 8; off <<= 1)
                tsum += __shfl_xor(tsum, off, 32);
            lrun[r] = lrun[r] * alpha + tsum;
            mrun[r] = mnew;
            #pragma unroll
            for (int j = 0; j < 4; ++j) oacc[j][r] *= alpha;
        }

        // ---- P: C-layout -> LDS -> A-fragments ----
        #pragma unroll
        for (int j = 0; j < 4; ++j)
            #pragma unroll
            for (int r = 0; r < 8; ++r)
                lP[w][(r + lh * 8) * 64 + j * 16 + ll] = (_Float16)sacc[j][r];
        __syncthreads();

        v16h pf[2];
        #pragma unroll
        for (int kk = 0; kk < 2; ++kk) {
            const _Float16* p = &lP[w][ll * 64 + kk * 32 + lh * 8];
            pf[kk] = ldfrag(p, p + 16);
        }

        // ---- O += P V (B-frags contiguous in transposed V), 8 wmma ----
        #pragma unroll
        for (int jd = 0; jd < 4; ++jd) {
            #pragma unroll
            for (int kk = 0; kk < 2; ++kk) {
                const _Float16* p = vptr + (size_t)(jd * 16 + ll) * S_
                                  + key0 + kk * 32 + lh * 8;
                v16h vf = ldfrag(p, p + 16);
                oacc[jd] = __builtin_amdgcn_wmma_f32_16x16x32_f16(
                    false, pf[kk], false, vf, (short)0, oacc[jd], false, false);
            }
        }
        __syncthreads();
    }

    // ---- normalize and write merged-head output [B,S,D] ----
    #pragma unroll
    for (int r = 0; r < 8; ++r) {
        float inv = 1.0f / lrun[r];
        int   s   = q0 + r + lh * 8;
        #pragma unroll
        for (int jd = 0; jd < 4; ++jd) {
            float v = oacc[jd][r] * inv;
            O[((size_t)b * S_ + s) * D_ + h * HD_ + jd * 16 + ll] = (_Float16)v;
        }
    }
}

// ---------------------------------------------------------------------------
// Host launcher. Workspace layout (f16 buffers):
//   [ 0,16MB)  Xh  (x + PE, [B,S,D])  -> reused as Oh after QKV
//   [16,32)    Qh  [B,H,S,64] (pre-scaled by 1/8)
//   [32,48)    Kh  [B,H,S,64]
//   [48,64)    Vt  [B,H,64,S]
//   [64,72)    WqH/WkH/WvH/WoH (f16 weights, 2 MB each)
// Total 72 MB of d_ws.
// ---------------------------------------------------------------------------
extern "C" void kernel_launch(void* const* d_in, const int* in_sizes, int n_in,
                              void* d_out, int out_size, void* d_ws, size_t ws_size,
                              hipStream_t stream) {
    const float* x  = (const float*)d_in[0];
    const float* Wq = (const float*)d_in[1];
    const float* bq = (const float*)d_in[2];
    const float* Wk = (const float*)d_in[3];
    const float* bk = (const float*)d_in[4];
    const float* Wv = (const float*)d_in[5];
    const float* bv = (const float*)d_in[6];
    const float* Wo = (const float*)d_in[7];
    const float* bo = (const float*)d_in[8];
    float* out = (float*)d_out;

    char* ws = (char*)d_ws;
    const size_t SZ = (size_t)B_ * S_ * D_ * sizeof(_Float16);  // 16 MB
    const size_t WZ = (size_t)D_ * D_ * sizeof(_Float16);       //  2 MB
    _Float16* Xh  = (_Float16*)(ws);
    _Float16* Qh  = (_Float16*)(ws + SZ);
    _Float16* Kh  = (_Float16*)(ws + 2 * SZ);
    _Float16* Vt  = (_Float16*)(ws + 3 * SZ);
    _Float16* WqH = (_Float16*)(ws + 4 * SZ);
    _Float16* WkH = (_Float16*)(ws + 4 * SZ + WZ);
    _Float16* WvH = (_Float16*)(ws + 4 * SZ + 2 * WZ);
    _Float16* WoH = (_Float16*)(ws + 4 * SZ + 3 * WZ);
    _Float16* Oh  = Xh;   // Xh dead after the QKV GEMMs

    const int wblocks = (D_ * D_) / (256 * 4);
    wcast_kernel<<<wblocks, 256, 0, stream>>>(Wq, WqH);
    wcast_kernel<<<wblocks, 256, 0, stream>>>(Wk, WkH);
    wcast_kernel<<<wblocks, 256, 0, stream>>>(Wv, WvH);
    wcast_kernel<<<wblocks, 256, 0, stream>>>(Wo, WoH);

    pe_cast_kernel<<<(B_ * S_ * D_) / 256, 256, 0, stream>>>(x, Xh);

    dim3 g(D_ / 128, M_ / 128);   // 8 x 64 blocks
    gemm_wmma_kernel<<<g, 256, 0, stream>>>(Xh, WqH, bq, Qh, nullptr, 0);
    gemm_wmma_kernel<<<g, 256, 0, stream>>>(Xh, WkH, bk, Kh, nullptr, 1);
    gemm_wmma_kernel<<<g, 256, 0, stream>>>(Xh, WvH, bv, Vt, nullptr, 2);

    attn_kernel<<<dim3(S_ / 64, H_, B_), 128, 0, stream>>>(Qh, Kh, Vt, Oh);

    gemm_wmma_kernel<<<g, 256, 0, stream>>>(Oh, WoH, bo, nullptr, out, 3);
}
